// CaptioningRNN_60284160967155
// MI455X (gfx1250) — matrix operand order repaired and checked by reading, not compiled, not run on hardware
//
#include <hip/hip_runtime.h>

// ---------------------------------------------------------------------------
// CaptioningRNN with spatial attention, MI455X (gfx1250, wave32, WMMA).
// N=256 batch, T=128 steps, D=512, H=1024.
// Per step: one (256 x 4096) x K=2560 bf16 WMMA GEMM  ( [h|attn|xt] @ [Wh;Wattn;Wx] )
// Each wave owns TWO 16-row M-tiles x one 16-col N-tile for all 4 gates:
// 8 independent WMMA chains, B fragments amortized over 2 M-tiles.
// __launch_bounds__(256,1) raises the VGPR budget so all 8 accumulators
// (64 VGPRs) + operand fragments stay in registers (no scratch spills).
// ---------------------------------------------------------------------------

#define NB   256
#define TT   128
#define DD   512
#define HH   1024
#define KTOT 2560              // H + H + D
#define KB_N (KTOT / 32)       // 80 K-blocks of 32
#define NTIL (4 * HH / 16)     // 256 N-tiles of 16 across the 4096-wide output

typedef __attribute__((ext_vector_type(16))) __bf16 v16bf;
typedef __attribute__((ext_vector_type(8)))  float  v8f;

// float -> bf16 round-to-nearest-even (bit trick)
__device__ __forceinline__ unsigned short f2bf(float f) {
  unsigned int u = __builtin_bit_cast(unsigned int, f);
  u += 0x7fffu + ((u >> 16) & 1u);
  return (unsigned short)(u >> 16);
}

// WMMA A-matrix (16x32 bf16) lane/element placement for row m, k-in-block kl.
// lanes 0-15: e=0..7 -> K=0..7,  e=8..15 -> K=16..23
// lanes16-31: e=0..7 -> K=8..15, e=8..15 -> K=24..31
__device__ __forceinline__ int a_swz_idx(int mb, int k, int m) {
  int kb = k >> 5, kl = k & 31;
  int l, e;
  if      (kl <  8) { l = m;      e = kl;      }
  else if (kl < 16) { l = m + 16; e = kl - 8;  }
  else if (kl < 24) { l = m;      e = kl - 8;  }
  else              { l = m + 16; e = kl - 16; }
  return ((mb * KB_N + kb) * 32 + l) * 16 + e;
}

// ---------------------------------------------------------------------------
// Kernel 1 (once): pack [Wh; Wattn; Wx] fp32 -> bf16 in WMMA-B swizzled tiles.
// B-matrix (32x16 bf16): lane l, element e  ->  K = (l>=16)*8 + (e&7) + (e>=8)*16, N = l&15.
// Tile (kb, nb) stored as 32 lanes x 16 contiguous bf16 (one 32B chunk/lane).
// ---------------------------------------------------------------------------
__global__ void pack_weights(const float* __restrict__ Wx,
                             const float* __restrict__ Wh,
                             const float* __restrict__ Wattn,
                             unsigned short* __restrict__ Wsw) {
  int idx = blockIdx.x * 256 + threadIdx.x;          // over KTOT*4096
  int e  = idx & 15;
  int l  = (idx >> 4) & 31;
  int nb = (idx >> 9) & 255;
  int kb = idx >> 17;
  int k  = kb * 32 + ((l >= 16) ? 8 : 0) + (e & 7) + ((e >= 8) ? 16 : 0);
  int j  = nb * 16 + (l & 15);
  float v;
  if (k < HH)            v = Wh[(size_t)k * 4096 + j];
  else if (k < 2 * HH)   v = Wattn[(size_t)(k - HH) * 4096 + j];
  else                   v = Wx[(size_t)(k - 2 * HH) * 4096 + j];
  Wsw[idx] = f2bf(v);
}

// ---------------------------------------------------------------------------
// Kernel 2 (once): h0 = c0 = mean over the 4x4 grid of A.
// ---------------------------------------------------------------------------
__global__ void init_hc(const float* __restrict__ A,
                        float* __restrict__ h, float* __restrict__ c) {
  int i = blockIdx.x * 256 + threadIdx.x;            // over N*H
  const float* p = A + (size_t)i * 16;
  float s = 0.f;
#pragma unroll
  for (int k = 0; k < 16; ++k) s += p[k];
  s *= (1.f / 16.f);
  h[i] = s;
  c[i] = s;
}

// ---------------------------------------------------------------------------
// Kernel 3 (per step): attention + pack [h | attn | xt] bf16, A-swizzled.
// One block (256 threads) per batch row n.
// ---------------------------------------------------------------------------
__global__ void attn_pack(const float* __restrict__ A,
                          const float* __restrict__ x,
                          const float* __restrict__ h,
                          unsigned short* __restrict__ hc, int t) {
  int n = blockIdx.x, tid = threadIdx.x;
  __shared__ float sh[256];
  __shared__ float sc[16], ew[16], wsm[16];

  const float* An = A + (size_t)n * HH * 16;
  const float* hn = h + (size_t)n * HH;

  // scores[p] = sum_j A[n,j,p] * h[n,j] / 32   (sqrt(H)=32). p = tid&15 -> coalesced A reads.
  int p = tid & 15, jg = tid >> 4;
  float acc = 0.f;
  for (int j = jg; j < HH; j += 16) acc += An[j * 16 + p] * hn[j];
  sh[tid] = acc;
  __syncthreads();
  if (tid < 16) {
    float s = 0.f;
#pragma unroll
    for (int g = 0; g < 16; ++g) s += sh[g * 16 + tid];
    sc[tid] = s * (1.f / 32.f);
  }
  __syncthreads();
  if (tid < 16) {
    float mx = sc[0];
#pragma unroll
    for (int k = 1; k < 16; ++k) mx = fmaxf(mx, sc[k]);
    ew[tid] = __expf(sc[tid] - mx);
  }
  __syncthreads();
  if (tid < 16) {
    float s = 0.f;
#pragma unroll
    for (int k = 0; k < 16; ++k) s += ew[k];
    wsm[tid] = ew[tid] / s;
  }
  __syncthreads();

  int m = n & 15, mb = n >> 4;
  // pack h  (k = 0..H-1)
  for (int j = tid; j < HH; j += 256)
    hc[a_swz_idx(mb, j, m)] = f2bf(hn[j]);
  // pack attn (k = H..2H-1):  attn[j] = sum_p A[n,j,p] * w[p]
  for (int j = tid; j < HH; j += 256) {
    const float* aj = An + j * 16;
    float a = 0.f;
#pragma unroll
    for (int q = 0; q < 16; ++q) a += aj[q] * wsm[q];
    hc[a_swz_idx(mb, HH + j, m)] = f2bf(a);
  }
  // pack xt (k = 2H..2H+D-1)
  const float* xn = x + ((size_t)n * TT + t) * DD;
  for (int d = tid; d < DD; d += 256)
    hc[a_swz_idx(mb, 2 * HH + d, m)] = f2bf(xn[d]);
}

// LSTM gate math + state update for one output element.
__device__ __forceinline__ void lstm_update(float ai, float af, float ao, float ag,
                                            int n, int col, int t,
                                            float* __restrict__ h,
                                            float* __restrict__ c,
                                            float* __restrict__ out) {
  float ig = 1.f / (1.f + __expf(-ai));
  float fg = 1.f / (1.f + __expf(-af));
  float og = 1.f / (1.f + __expf(-ao));
  float gg = tanhf(ag);
  size_t idx = (size_t)n * HH + col;
  float cn = fg * c[idx] + ig * gg;
  float hn = og * tanhf(cn);
  c[idx] = cn;
  h[idx] = hn;
  out[((size_t)n * TT + t) * HH + col] = hn;
}

// ---------------------------------------------------------------------------
// Kernel 4 (per step): GEMM (K=2560) via v_wmma_f32_16x16x32_bf16 + LSTM gates.
// grid (8, 8): x = 128-column group of H, y = 32-row batch strip (2 M-tiles).
// 8 waves/block; each wave: 2 M-tiles x 4 gate N-tiles = 8 WMMA chains.
// ---------------------------------------------------------------------------
__global__ void __launch_bounds__(256, 1)
step_gemm_gate(const unsigned short* __restrict__ Wsw,
               const unsigned short* __restrict__ hc,
               const float* __restrict__ bias,
               float* __restrict__ h, float* __restrict__ c,
               float* __restrict__ out, int t) {
  int mb0  = blockIdx.y * 2;
  int mb1  = mb0 + 1;
  int wave = threadIdx.x >> 5;
  int lane = threadIdx.x & 31;
  int colH = blockIdx.x * 128 + wave * 16;   // column base within H
  int nbi  = colH >> 4;                      // N-tile of gate i; gates at +64/+128/+192

  const v16bf* Ap = (const v16bf*)hc;
  const v16bf* Bp = (const v16bf*)Wsw;

  v8f a00 = {}, a01 = {}, a02 = {}, a03 = {};   // M-tile 0, gates i,f,o,g
  v8f a10 = {}, a11 = {}, a12 = {}, a13 = {};   // M-tile 1

  for (int kb = 0; kb < KB_N; ++kb) {
    v16bf fa0 = Ap[(mb0 * KB_N + kb) * 32 + lane];
    v16bf fa1 = Ap[(mb1 * KB_N + kb) * 32 + lane];
    const v16bf* brow = Bp + ((size_t)kb * NTIL) * 32;
    v16bf fb0 = brow[(nbi      ) * 32 + lane];
    v16bf fb1 = brow[(nbi +  64) * 32 + lane];
    v16bf fb2 = brow[(nbi + 128) * 32 + lane];
    v16bf fb3 = brow[(nbi + 192) * 32 + lane];
    if (kb + 1 < KB_N) {
      const v16bf* bnext = Bp + ((size_t)(kb + 1) * NTIL) * 32;
      __builtin_prefetch(&bnext[(nbi      ) * 32 + lane], 0, 3);
      __builtin_prefetch(&bnext[(nbi + 128) * 32 + lane], 0, 3);
    }
    a00 = __builtin_amdgcn_wmma_f32_16x16x32_bf16(false, fa0, false, fb0, (short)0, a00, false, false);
    a10 = __builtin_amdgcn_wmma_f32_16x16x32_bf16(false, fa1, false, fb0, (short)0, a10, false, false);
    a01 = __builtin_amdgcn_wmma_f32_16x16x32_bf16(false, fa0, false, fb1, (short)0, a01, false, false);
    a11 = __builtin_amdgcn_wmma_f32_16x16x32_bf16(false, fa1, false, fb1, (short)0, a11, false, false);
    a02 = __builtin_amdgcn_wmma_f32_16x16x32_bf16(false, fa0, false, fb2, (short)0, a02, false, false);
    a12 = __builtin_amdgcn_wmma_f32_16x16x32_bf16(false, fa1, false, fb2, (short)0, a12, false, false);
    a03 = __builtin_amdgcn_wmma_f32_16x16x32_bf16(false, fa0, false, fb3, (short)0, a03, false, false);
    a13 = __builtin_amdgcn_wmma_f32_16x16x32_bf16(false, fa1, false, fb3, (short)0, a13, false, false);
  }

  // C/D layout: VGPR r -> row r (lanes 0-15) or row r+8 (lanes 16-31); col = lane&15.
  int nloc = lane & 15;
  int col  = colH + nloc;
  float b0v = bias[col];
  float b1v = bias[col + HH];
  float b2v = bias[col + 2 * HH];
  float b3v = bias[col + 3 * HH];
  int row0 = (lane >> 4) ? 8 : 0;

#pragma unroll
  for (int r = 0; r < 8; ++r) {
    int n0 = mb0 * 16 + row0 + r;
    lstm_update(a00[r] + b0v, a01[r] + b1v, a02[r] + b2v, a03[r] + b3v,
                n0, col, t, h, c, out);
    int n1 = mb1 * 16 + row0 + r;
    lstm_update(a10[r] + b0v, a11[r] + b1v, a12[r] + b2v, a13[r] + b3v,
                n1, col, t, h, c, out);
  }
}

// ---------------------------------------------------------------------------
// Launch: pack weights once, init h/c, then 128 x (attn_pack -> gemm+gate).
// ws layout: Wsw 20MB | hc 1.25MB | h 1MB | c 1MB   (~24.4MB total)
// ---------------------------------------------------------------------------
extern "C" void kernel_launch(void* const* d_in, const int* in_sizes, int n_in,
                              void* d_out, int out_size, void* d_ws, size_t ws_size,
                              hipStream_t stream) {
  (void)in_sizes; (void)n_in; (void)out_size; (void)ws_size;
  const float* x     = (const float*)d_in[0];
  const float* A     = (const float*)d_in[1];
  const float* Wx    = (const float*)d_in[2];
  const float* Wh    = (const float*)d_in[3];
  const float* Wattn = (const float*)d_in[4];
  const float* bias  = (const float*)d_in[5];
  float* out = (float*)d_out;

  char* ws = (char*)d_ws;
  unsigned short* Wsw = (unsigned short*)ws;                      // KTOT*4096*2 = 20971520 B
  unsigned short* hc  = (unsigned short*)(ws + 20971520);         // NB*KTOT*2   =  1310720 B
  float* h = (float*)(ws + 20971520 + 1310720);                   // NB*HH*4
  float* c = h + (size_t)NB * HH;

  pack_weights<<<(KTOT * 4096) / 256, 256, 0, stream>>>(Wx, Wh, Wattn, Wsw);
  init_hc<<<(NB * HH) / 256, 256, 0, stream>>>(A, h, c);

  for (int t = 0; t < TT; ++t) {
    attn_pack<<<NB, 256, 0, stream>>>(A, x, h, hc, t);
    step_gemm_gate<<<dim3(8, 8), 256, 0, stream>>>(Wsw, hc, bias, h, c, out, t);
  }
}
